// MMDiTBlock_from_DiTBlock_75222057223010
// MI455X (gfx1250) — matrix operand order repaired and checked by reading, not compiled, not run on hardware
//
#include <hip/hip_runtime.h>
#include <hip/hip_bf16.h>
#include <math.h>

typedef __bf16 bf16;
typedef __attribute__((ext_vector_type(16))) __bf16 v16bf;
typedef __attribute__((ext_vector_type(8)))  float  v8f;

#define B_DIM   4
#define N1_DIM  1024
#define N2_DIM  128
#define NT_DIM  1152
#define C_DIM   1152
#define H_DIM   16
#define D_DIM   72
#define DP_QK   96
#define DP_V    80
#define HID_DIM 4608
#define SIXC    6912
#define ATTN_SCALE_F 0.11785113019775793f
#define RMS_SCALE_F  33.941125496954285f

struct alignas(16) U16B { unsigned int x0, x1, x2, x3; };
union Frag { U16B q[2]; v16bf v; };

__device__ __forceinline__ float b2f(bf16 x) { return (float)x; }
__device__ __forceinline__ bf16  f2b(float x) { return (bf16)x; }

// ---------------------------------------------------------------------------
// fp32 (K x N) -> bf16 transposed (N x K), LDS-tiled 32x32
// ---------------------------------------------------------------------------
__global__ __launch_bounds__(256) void cvt_f2b_t(const float* __restrict__ in,
                                                 bf16* __restrict__ out,
                                                 int K, int N) {
    __shared__ bf16 tile[32][33];
    const int tx = threadIdx.x & 31, ty = threadIdx.x >> 5;  // 32 x 8
    const int k0 = blockIdx.y * 32, n0 = blockIdx.x * 32;
    for (int i = ty; i < 32; i += 8)
        tile[i][tx] = f2b(in[(long)(k0 + i) * N + n0 + tx]);
    __syncthreads();
    for (int i = ty; i < 32; i += 8)
        out[(long)(n0 + i) * K + k0 + tx] = tile[tx][i];
}

// ---------------------------------------------------------------------------
// adaLN modulation: mod = silu(global_c) @ ada_w + ada_b   (both x and c)
// ---------------------------------------------------------------------------
__global__ __launch_bounds__(256) void ada_mod(const float* __restrict__ gc,
                                               const float* __restrict__ wx,
                                               const float* __restrict__ bx,
                                               const float* __restrict__ wc,
                                               const float* __restrict__ bc,
                                               float* __restrict__ mod_x,
                                               float* __restrict__ mod_c) {
    const int idx = blockIdx.x * 256 + threadIdx.x;
    if (idx >= 2 * B_DIM * SIXC) return;
    const int which = idx / (B_DIM * SIXC);
    const int r     = idx % (B_DIM * SIXC);
    const int b     = r / SIXC, col = r % SIXC;
    const float* wgt = which ? wc : wx;
    const float* bb  = which ? bc : bx;
    float acc = 0.f;
    for (int j = 0; j < C_DIM; ++j) {
        float s = gc[b * C_DIM + j];
        s = s / (1.f + __expf(-s));
        acc += s * wgt[(long)j * SIXC + col];
    }
    (which ? mod_c : mod_x)[b * SIXC + col] = acc + bb[col];
}

// ---------------------------------------------------------------------------
// LayerNorm + modulate, writes bf16 rows
// ---------------------------------------------------------------------------
__global__ __launch_bounds__(256) void ln_mod(const float* __restrict__ in,
                                              const float* __restrict__ g,
                                              const float* __restrict__ be,
                                              const float* __restrict__ mod,
                                              int shOff, int scOff, int rows_per_b,
                                              bf16* __restrict__ out) {
    __shared__ float s1[256], s2[256];
    const int row = blockIdx.x, tid = threadIdx.x;
    const int b = row / rows_per_b;
    const float* x = in + (long)row * C_DIM;
    float p1 = 0.f, p2 = 0.f;
    for (int j = tid; j < C_DIM; j += 256) { float v = x[j]; p1 += v; p2 += v * v; }
    s1[tid] = p1; s2[tid] = p2; __syncthreads();
    for (int off = 128; off; off >>= 1) {
        if (tid < off) { s1[tid] += s1[tid + off]; s2[tid] += s2[tid + off]; }
        __syncthreads();
    }
    const float mean = s1[0] * (1.f / C_DIM);
    const float var  = s2[0] * (1.f / C_DIM) - mean * mean;
    const float rstd = rsqrtf(var + 1e-6f);
    const float* mrow = mod + (long)b * SIXC;
    bf16* o = out + (long)row * C_DIM;
    for (int j = tid; j < C_DIM; j += 256) {
        float v = (x[j] - mean) * rstd * g[j] + be[j];
        o[j] = f2b(v * (1.f + mrow[scOff + j]) + mrow[shOff + j]);
    }
}

// ---------------------------------------------------------------------------
// bf16 WMMA GEMM: D = A @ W + bias.  W given pre-TRANSPOSED (N x K row-major),
// so each B fragment is two contiguous 16B per-lane global loads (L2-resident).
// A tile double-buffered in LDS; one barrier per K-step.
//   mode 0: outF = acc+bias
//   mode 1: outB = bf16(gelu_tanh(acc+bias))
//   mode 2: outF = res + gate[b,col]*(acc+bias)
//   mode 3: outB = bf16(acc+bias)
// Row map (reads packed attention output): arow = (r/a_rpb)*a_sb + a_off + r%a_rpb
// ---------------------------------------------------------------------------
__global__ __launch_bounds__(256) void gemm_bf16(
    const bf16* __restrict__ A, int a_rpb, int a_sb, int a_off, int lda,
    const bf16* __restrict__ Wt, int ldk,
    const float* __restrict__ bias,
    float* __restrict__ outF, bf16* __restrict__ outB,
    const float* __restrict__ res, const float* __restrict__ gate,
    int gstride, int out_rpb,
    int M, int N, int K, int mode) {
    __shared__ __align__(16) bf16 As[2][128 * 40];
    const int tid = threadIdx.x;
    const int w = tid >> 5, lane = tid & 31;
    const int wy = w & 1, wx = w >> 1;
    const int r0 = blockIdx.y * 128, n0 = blockIdx.x * 128;

    const v8f vzero = {0.f, 0.f, 0.f, 0.f, 0.f, 0.f, 0.f, 0.f};
    v8f acc[4][2];
    for (int i = 0; i < 4; ++i) for (int j = 0; j < 2; ++j) acc[i][j] = vzero;

    // A staging: each thread moves 16 contiguous bf16 of one tile row
    const int sa_row = tid >> 1;
    const int sa_c   = (tid & 1) << 4;
    const int ar     = r0 + sa_row;
    const long agrow = (long)(ar / a_rpb) * a_sb + a_off + (ar % a_rpb);
    const bf16* aptr = A + agrow * (long)lda + sa_c;

    const int fl  = lane & 15;
    const int ahi = (lane >> 4) << 3;  // 0 or 8
    const int bkh = (lane >> 4) << 4;  // 0 or 16
    // per-lane B row bases (pre-transposed weights)
    const bf16* bptr0 = Wt + (long)(n0 + wx * 32 + fl) * ldk + ahi;
    const bf16* bptr1 = bptr0 + (long)16 * ldk;

    const int ksteps = K >> 5;
    // prologue: stage k-step 0
    {
        const U16B* src = (const U16B*)(aptr);
        U16B v0 = src[0], v1 = src[1];
        U16B* dst = (U16B*)(&As[0][sa_row * 40 + sa_c]);
        dst[0] = v0; dst[1] = v1;
    }
    __syncthreads();

    for (int kt = 0; kt < ksteps; ++kt) {
        const int k0 = kt << 5;
        const int cur = kt & 1;
        // stage next A tile into the other buffer
        if (kt + 1 < ksteps) {
            const U16B* src = (const U16B*)(aptr + k0 + 32);
            U16B v0 = src[0], v1 = src[1];
            U16B* dst = (U16B*)(&As[cur ^ 1][sa_row * 40 + sa_c]);
            dst[0] = v0; dst[1] = v1;
            __builtin_prefetch(aptr + k0 + 64, 0, 1);
            __builtin_prefetch(bptr0 + k0 + 32, 0, 1);
        }
        // A fragments from LDS
        Frag afr[4];
#pragma unroll
        for (int mb = 0; mb < 4; ++mb) {
            const int rr = wy * 64 + mb * 16 + fl;
            afr[mb].q[0] = *(const U16B*)(&As[cur][rr * 40 + ahi]);
            afr[mb].q[1] = *(const U16B*)(&As[cur][rr * 40 + 16 + ahi]);
        }
        // B fragments direct from global (two 16B loads per lane)
        Frag bfr[2];
        bfr[0].q[0] = *(const U16B*)(bptr0 + k0);
        bfr[0].q[1] = *(const U16B*)(bptr0 + k0 + 16);
        bfr[1].q[0] = *(const U16B*)(bptr1 + k0);
        bfr[1].q[1] = *(const U16B*)(bptr1 + k0 + 16);
#pragma unroll
        for (int mb = 0; mb < 4; ++mb)
#pragma unroll
            for (int nb = 0; nb < 2; ++nb)
                acc[mb][nb] = __builtin_amdgcn_wmma_f32_16x16x32_bf16(
                    false, afr[mb].v, false, bfr[nb].v, (short)0, acc[mb][nb], false, false);
        __syncthreads();
    }

    const int m0 = (lane < 16) ? 0 : 8;
#pragma unroll
    for (int mb = 0; mb < 4; ++mb) {
#pragma unroll
        for (int nb = 0; nb < 2; ++nb) {
            const int gcol = n0 + wx * 32 + nb * 16 + fl;
            const float bv = bias ? bias[gcol] : 0.f;
#pragma unroll
            for (int r = 0; r < 8; ++r) {
                const int grow = r0 + wy * 64 + mb * 16 + m0 + r;
                float v = acc[mb][nb][r] + bv;
                const long oidx = (long)grow * N + gcol;
                if (mode == 0) {
                    outF[oidx] = v;
                } else if (mode == 1) {
                    float gl = 0.5f * v * (1.f + tanhf(0.7978845608f * (v + 0.044715f * v * v * v)));
                    outB[oidx] = f2b(gl);
                } else if (mode == 2) {
                    const float gt = gate[(long)(grow / out_rpb) * gstride + gcol];
                    outF[oidx] = res[oidx] + gt * v;
                } else {
                    outB[oidx] = f2b(v);
                }
            }
        }
    }
}

// ---------------------------------------------------------------------------
// qkv post-process: split heads, RMS-norm q/k (scale sqrt(C)*g), write q/k
// padded to Dp=96, and V transposed to (b,h,d,t) padded to 80 rows.
// ---------------------------------------------------------------------------
__global__ __launch_bounds__(256) void qkv_post(
    const bf16* __restrict__ qkvx, const bf16* __restrict__ qkvc,
    const float* __restrict__ gqx, const float* __restrict__ gkx,
    const float* __restrict__ gqc, const float* __restrict__ gkc,
    bf16* __restrict__ q, bf16* __restrict__ k, bf16* __restrict__ vT) {
    const int idx = blockIdx.x * 256 + threadIdx.x;
    if (idx >= B_DIM * H_DIM * NT_DIM) return;
    const int t  = idx % NT_DIM;
    const int bh = idx / NT_DIM;
    const int b = bh >> 4, h = bh & 15;
    const bf16* src; float gq, gk;
    if (t < N1_DIM) { src = qkvx + ((long)(b * N1_DIM + t)) * (3 * C_DIM); gq = gqx[0]; gk = gkx[0]; }
    else            { src = qkvc + ((long)(b * N2_DIM + (t - N1_DIM))) * (3 * C_DIM); gq = gqc[0]; gk = gkc[0]; }
    float sq = 0.f, sk = 0.f;
    for (int d = 0; d < D_DIM; ++d) {
        float a  = b2f(src[h * D_DIM + d]);           sq += a * a;
        float c2 = b2f(src[C_DIM + h * D_DIM + d]);   sk += c2 * c2;
    }
    const float qs  = RMS_SCALE_F * gq / fmaxf(sqrtf(sq), 1e-12f);
    const float ks_ = RMS_SCALE_F * gk / fmaxf(sqrtf(sk), 1e-12f);
    const long qb = ((long)bh * NT_DIM + t) * DP_QK;
    for (int d = 0; d < D_DIM; ++d) {
        q[qb + d] = f2b(b2f(src[h * D_DIM + d]) * qs);
        k[qb + d] = f2b(b2f(src[C_DIM + h * D_DIM + d]) * ks_);
    }
    for (int d = D_DIM; d < DP_QK; ++d) { q[qb + d] = f2b(0.f); k[qb + d] = f2b(0.f); }
    for (int d = 0; d < D_DIM; ++d)
        vT[((long)bh * DP_V + d) * NT_DIM + t] = src[2 * C_DIM + h * D_DIM + d];
    for (int d = D_DIM; d < DP_V; ++d)
        vT[((long)bh * DP_V + d) * NT_DIM + t] = f2b(0.f);
}

// ---------------------------------------------------------------------------
// Attention: per (b,h, 16-row Q tile). S=QK^T (WMMA, Dp=96), softmax in LDS,
// O=PV (WMMA, K=1152 split over 8 waves + LDS reduce). Output packed (B,NT,C) bf16.
// ---------------------------------------------------------------------------
__global__ __launch_bounds__(256) void attn(const bf16* __restrict__ q,
                                            const bf16* __restrict__ k,
                                            const bf16* __restrict__ vT,
                                            bf16* __restrict__ obuf) {
    __shared__ __align__(16) float S[16 * 1160];
    __shared__ __align__(16) bf16  P[16 * 1160];
    __shared__ float rsum[16];
    const int tid = threadIdx.x;
    const int w = tid >> 5, lane = tid & 31;
    const int bh = blockIdx.x, qt = blockIdx.y;
    const int b = bh >> 4, h = bh & 15;
    const int fl  = lane & 15;
    const int ahi = (lane >> 4) << 3;
    const int bkh = (lane >> 4) << 4;

    Frag qf[3];
    const long qrow = ((long)bh * NT_DIM + qt * 16 + fl) * DP_QK;
#pragma unroll
    for (int kb = 0; kb < 3; ++kb) {
        const bf16* qp = q + qrow + kb * 32 + ahi;
        qf[kb].q[0] = *(const U16B*)qp;
        qf[kb].q[1] = *(const U16B*)(qp + 16);
    }
    const v8f vzero = {0.f, 0.f, 0.f, 0.f, 0.f, 0.f, 0.f, 0.f};

    for (int cb = w; cb < 72; cb += 8) {
        v8f s = vzero;
        const long krow = ((long)bh * NT_DIM + cb * 16 + fl) * DP_QK;
#pragma unroll
        for (int kb = 0; kb < 3; ++kb) {
            Frag kf;
            const bf16* kp = k + krow + kb * 32 + ahi;
            kf.q[0] = *(const U16B*)kp;
            kf.q[1] = *(const U16B*)(kp + 16);
            s = __builtin_amdgcn_wmma_f32_16x16x32_bf16(false, qf[kb].v, false, kf.v,
                                                        (short)0, s, false, false);
        }
        const int m0 = (lane < 16) ? 0 : 8;
#pragma unroll
        for (int r = 0; r < 8; ++r)
            S[(m0 + r) * 1160 + cb * 16 + fl] = s[r] * ATTN_SCALE_F;
    }
    __syncthreads();

    const int row = tid >> 4, t16 = tid & 15;
    float mx = -3.0e38f;
    for (int j = t16; j < NT_DIM; j += 16) mx = fmaxf(mx, S[row * 1160 + j]);
    for (int off = 8; off; off >>= 1) mx = fmaxf(mx, __shfl_xor(mx, off, 16));
    float sum = 0.f;
    for (int j = t16; j < NT_DIM; j += 16) {
        float e = __expf(S[row * 1160 + j] - mx);
        P[row * 1160 + j] = f2b(e);
        sum += e;
    }
    for (int off = 8; off; off >>= 1) sum += __shfl_xor(sum, off, 16);
    if (t16 == 0) rsum[row] = sum;
    __syncthreads();

    v8f oacc[5];
    for (int nb = 0; nb < 5; ++nb) oacc[nb] = vzero;
    for (int ks = w; ks < 36; ks += 8) {
        const int Kb = ks * 32;
        Frag pf;
        const bf16* pp = &P[fl * 1160 + Kb + ahi];
        pf.q[0] = *(const U16B*)pp;
        pf.q[1] = *(const U16B*)(pp + 16);
#pragma unroll
        for (int nb = 0; nb < 5; ++nb) {
            Frag vf;
            const bf16* vp = vT + ((long)bh * DP_V + nb * 16 + fl) * NT_DIM + Kb + ahi;
            vf.q[0] = *(const U16B*)vp;
            vf.q[1] = *(const U16B*)(vp + 16);
            oacc[nb] = __builtin_amdgcn_wmma_f32_16x16x32_bf16(false, pf.v, false, vf.v,
                                                               (short)0, oacc[nb], false, false);
        }
    }
    float* Ored = S;  // S no longer needed
    const int m0 = (lane < 16) ? 0 : 8;
#pragma unroll
    for (int nb = 0; nb < 5; ++nb)
#pragma unroll
        for (int r = 0; r < 8; ++r)
            Ored[(w * 16 + m0 + r) * 80 + nb * 16 + fl] = oacc[nb][r];
    __syncthreads();
    for (int idx = tid; idx < 16 * 80; idx += 256) {
        const int m = idx / 80, n = idx % 80;
        float s = 0.f;
#pragma unroll
        for (int ww = 0; ww < 8; ++ww) s += Ored[(ww * 16 + m) * 80 + n];
        if (n < D_DIM) {
            const float inv = 1.f / fmaxf(rsum[m], 1e-30f);
            obuf[((long)b * NT_DIM + qt * 16 + m) * C_DIM + h * D_DIM + n] = f2b(s * inv);
        }
    }
}

// ---------------------------------------------------------------------------
extern "C" void kernel_launch(void* const* d_in, const int* in_sizes, int n_in,
                              void* d_out, int out_size, void* d_ws, size_t ws_size,
                              hipStream_t stream) {
    (void)in_sizes; (void)n_in; (void)out_size; (void)ws_size;
    const float* x        = (const float*)d_in[0];
    const float* c        = (const float*)d_in[1];
    const float* global_c = (const float*)d_in[2];
    const float* ada_x_w  = (const float*)d_in[3];
    const float* ada_x_b  = (const float*)d_in[4];
    const float* ada_c_w  = (const float*)d_in[5];
    const float* ada_c_b  = (const float*)d_in[6];
    const float* g1x_g = (const float*)d_in[7];
    const float* g1x_b = (const float*)d_in[8];
    const float* g2x_g = (const float*)d_in[9];
    const float* g2x_b = (const float*)d_in[10];
    const float* g1c_g = (const float*)d_in[11];
    const float* g1c_b = (const float*)d_in[12];
    const float* g2c_g = (const float*)d_in[13];
    const float* g2c_b = (const float*)d_in[14];
    const float* qkv_x_w  = (const float*)d_in[15];
    const float* qkv_x_b  = (const float*)d_in[16];
    const float* proj_x_w = (const float*)d_in[17];
    const float* proj_x_b = (const float*)d_in[18];
    const float* qkv_c_w  = (const float*)d_in[19];
    const float* qkv_c_b  = (const float*)d_in[20];
    const float* proj_c_w = (const float*)d_in[21];
    const float* proj_c_b = (const float*)d_in[22];
    const float* gq_x = (const float*)d_in[23];
    const float* gk_x = (const float*)d_in[24];
    const float* gq_c = (const float*)d_in[25];
    const float* gk_c = (const float*)d_in[26];
    const float* mlp_x_w1 = (const float*)d_in[27];
    const float* mlp_x_b1 = (const float*)d_in[28];
    const float* mlp_x_w2 = (const float*)d_in[29];
    const float* mlp_x_b2 = (const float*)d_in[30];
    const float* mlp_c_w1 = (const float*)d_in[31];
    const float* mlp_c_b1 = (const float*)d_in[32];
    const float* mlp_c_w2 = (const float*)d_in[33];
    const float* mlp_c_b2 = (const float*)d_in[34];

    float* outx = (float*)d_out;
    float* outc = outx + (size_t)B_DIM * N1_DIM * C_DIM;

    char* ws = (char*)d_ws;
    size_t off = 0;
    auto alloc = [&](size_t bytes) -> char* {
        char* p = ws + off;
        off += (bytes + 255) & ~(size_t)255;
        return p;
    };
    const int Mx = B_DIM * N1_DIM;  // 4096
    const int Mc = B_DIM * N2_DIM;  // 512
    float* modx = (float*)alloc((size_t)B_DIM * SIXC * 4);
    float* modc = (float*)alloc((size_t)B_DIM * SIXC * 4);
    bf16* xin   = (bf16*)alloc((size_t)Mx * C_DIM * 2);
    bf16* cin   = (bf16*)alloc((size_t)Mc * C_DIM * 2);
    bf16* wqkvx = (bf16*)alloc((size_t)C_DIM * 3 * C_DIM * 2);   // transposed 3C x C
    bf16* wqkvc = (bf16*)alloc((size_t)C_DIM * 3 * C_DIM * 2);
    bf16* wprjx = (bf16*)alloc((size_t)C_DIM * C_DIM * 2);
    bf16* wprjc = (bf16*)alloc((size_t)C_DIM * C_DIM * 2);
    bf16* wm1x  = (bf16*)alloc((size_t)C_DIM * HID_DIM * 2);     // HID x C
    bf16* wm2x  = (bf16*)alloc((size_t)HID_DIM * C_DIM * 2);     // C x HID
    bf16* wm1c  = (bf16*)alloc((size_t)C_DIM * HID_DIM * 2);
    bf16* wm2c  = (bf16*)alloc((size_t)HID_DIM * C_DIM * 2);
    bf16* qkvx  = (bf16*)alloc((size_t)Mx * 3 * C_DIM * 2);
    bf16* qkvc  = (bf16*)alloc((size_t)Mc * 3 * C_DIM * 2);
    bf16* qb    = (bf16*)alloc((size_t)B_DIM * H_DIM * NT_DIM * DP_QK * 2);
    bf16* kb    = (bf16*)alloc((size_t)B_DIM * H_DIM * NT_DIM * DP_QK * 2);
    bf16* vTb   = (bf16*)alloc((size_t)B_DIM * H_DIM * DP_V * NT_DIM * 2);
    bf16* obuf  = (bf16*)alloc((size_t)B_DIM * NT_DIM * C_DIM * 2);
    bf16* hbuf  = (bf16*)alloc((size_t)Mx * HID_DIM * 2);

    // convert + transpose weights: in is K x N, out is N x K bf16
    auto cvtT = [&](const float* src, bf16* dst, int K, int N) {
        cvt_f2b_t<<<dim3(N / 32, K / 32), dim3(256), 0, stream>>>(src, dst, K, N);
    };
    cvtT(qkv_x_w, wqkvx, C_DIM, 3 * C_DIM);
    cvtT(qkv_c_w, wqkvc, C_DIM, 3 * C_DIM);
    cvtT(proj_x_w, wprjx, C_DIM, C_DIM);
    cvtT(proj_c_w, wprjc, C_DIM, C_DIM);
    cvtT(mlp_x_w1, wm1x, C_DIM, HID_DIM);
    cvtT(mlp_x_w2, wm2x, HID_DIM, C_DIM);
    cvtT(mlp_c_w1, wm1c, C_DIM, HID_DIM);
    cvtT(mlp_c_w2, wm2c, HID_DIM, C_DIM);

    ada_mod<<<dim3((2 * B_DIM * SIXC + 255) / 256), dim3(256), 0, stream>>>(
        global_c, ada_x_w, ada_x_b, ada_c_w, ada_c_b, modx, modc);

    ln_mod<<<dim3(Mx), dim3(256), 0, stream>>>(x, g1x_g, g1x_b, modx, 0, C_DIM, N1_DIM, xin);
    ln_mod<<<dim3(Mc), dim3(256), 0, stream>>>(c, g1c_g, g1c_b, modc, 0, C_DIM, N2_DIM, cin);

    auto gemm = [&](const bf16* A, int a_rpb, int a_sb, int a_off, int lda,
                    const bf16* Wt, int ldk, const float* bias,
                    float* oF, bf16* oB, const float* res, const float* gate,
                    int out_rpb, int M, int N, int K, int mode) {
        gemm_bf16<<<dim3(N / 128, M / 128), dim3(256), 0, stream>>>(
            A, a_rpb, a_sb, a_off, lda, Wt, ldk, bias, oF, oB, res, gate,
            SIXC, out_rpb, M, N, K, mode);
    };

    // qkv projections -> bf16
    gemm(xin, Mx, 0, 0, C_DIM, wqkvx, C_DIM, qkv_x_b, nullptr, qkvx,
         nullptr, nullptr, 1, Mx, 3 * C_DIM, C_DIM, 3);
    gemm(cin, Mc, 0, 0, C_DIM, wqkvc, C_DIM, qkv_c_b, nullptr, qkvc,
         nullptr, nullptr, 1, Mc, 3 * C_DIM, C_DIM, 3);

    qkv_post<<<dim3((B_DIM * H_DIM * NT_DIM + 255) / 256), dim3(256), 0, stream>>>(
        qkvx, qkvc, gq_x, gk_x, gq_c, gk_c, qb, kb, vTb);

    attn<<<dim3(B_DIM * H_DIM, NT_DIM / 16), dim3(256), 0, stream>>>(qb, kb, vTb, obuf);

    // attention output projection + gated residual (reads packed obuf via row map)
    gemm(obuf, N1_DIM, NT_DIM, 0, C_DIM, wprjx, C_DIM, proj_x_b, outx, nullptr,
         x, modx + 2 * C_DIM, N1_DIM, Mx, C_DIM, C_DIM, 2);
    gemm(obuf, N2_DIM, NT_DIM, N1_DIM, C_DIM, wprjc, C_DIM, proj_c_b, outc, nullptr,
         c, modc + 2 * C_DIM, N2_DIM, Mc, C_DIM, C_DIM, 2);

    // MLP for x
    ln_mod<<<dim3(Mx), dim3(256), 0, stream>>>(outx, g2x_g, g2x_b, modx,
                                               3 * C_DIM, 4 * C_DIM, N1_DIM, xin);
    gemm(xin, Mx, 0, 0, C_DIM, wm1x, C_DIM, mlp_x_b1, nullptr, hbuf,
         nullptr, nullptr, 1, Mx, HID_DIM, C_DIM, 1);
    gemm(hbuf, Mx, 0, 0, HID_DIM, wm2x, HID_DIM, mlp_x_b2, outx, nullptr,
         outx, modx + 5 * C_DIM, N1_DIM, Mx, C_DIM, HID_DIM, 2);

    // MLP for c
    ln_mod<<<dim3(Mc), dim3(256), 0, stream>>>(outc, g2c_g, g2c_b, modc,
                                               3 * C_DIM, 4 * C_DIM, N2_DIM, cin);
    gemm(cin, Mc, 0, 0, C_DIM, wm1c, C_DIM, mlp_c_b1, nullptr, hbuf,
         nullptr, nullptr, 1, Mc, HID_DIM, C_DIM, 1);
    gemm(hbuf, Mc, 0, 0, HID_DIM, wm2c, HID_DIM, mlp_c_b2, outc, nullptr,
         outc, modc + 5 * C_DIM, N2_DIM, Mc, C_DIM, HID_DIM, 2);
}
